// Lfm2MonolithicWrapper_53901839565591
// MI455X (gfx1250) — matrix operand order repaired and checked by reading, not compiled, not run on hardware
//
#include <hip/hip_runtime.h>
#include <hip/hip_bf16.h>
#include <math.h>

#define H 1024
#define NH 16
#define NKV 8
#define HD 64
#define CTX 2048
#define VOCAB 32768
#define DFF 4096
#define NL 12
#define NA 4
#define NC 8

typedef __attribute__((ext_vector_type(16))) _Float16 v16h;
typedef __attribute__((ext_vector_type(8)))  float    v8f;
typedef __attribute__((ext_vector_type(4)))  int      v4i;

// Pointer types matching this toolchain's async-LDS builtin signature:
// param0 = AS1 (global) v4i*, param1 = AS3 (LDS) v4i*
typedef __attribute__((address_space(1))) v4i GA4;
typedef __attribute__((address_space(3))) v4i LA4;

#if defined(__has_builtin)
#if __has_builtin(__builtin_amdgcn_global_load_async_to_lds_b128)
#define HAVE_ASYNC_LDS 1
#endif
#if __has_builtin(__builtin_amdgcn_s_wait_asynccnt)
#define HAVE_WAIT_ASYNC 1
#endif
#endif

__device__ __forceinline__ float waveReduceSum(float v) {
  #pragma unroll
  for (int off = 16; off; off >>= 1) v += __shfl_xor(v, off, 32);
  return v;
}

// Stage n4 float4's of x into LDS, using the gfx1250 async Global->LDS path
// when the toolchain exposes it (ASYNCcnt-tracked, no VGPR roundtrip).
__device__ __forceinline__ void stage_x(const float* __restrict__ x,
                                        float* xs, int n4, int tid) {
#if defined(HAVE_ASYNC_LDS)
  for (int i = tid; i < n4; i += 256)
    __builtin_amdgcn_global_load_async_to_lds_b128(
        (GA4*)(x + 4 * i), (LA4*)(xs + 4 * i), 0, 0);
#if defined(HAVE_WAIT_ASYNC)
  __builtin_amdgcn_s_wait_asynccnt(0);
#else
  asm volatile("s_wait_asynccnt 0x0" ::: "memory");
#endif
#else
  for (int i = tid; i < n4; i += 256)
    ((float4*)xs)[i] = ((const float4*)x)[i];
#endif
  __syncthreads();
}

// ---------------------------------------------------------------------------
// Embedding lookup: hidden = embed_w[id]
// ---------------------------------------------------------------------------
__global__ void embed_kernel(const int* __restrict__ ids,
                             const float* __restrict__ E,
                             float* __restrict__ h) {
  int i = blockIdx.x * blockDim.x + threadIdx.x;
  if (i < H) h[i] = E[(size_t)ids[0] * H + i];
}

// ---------------------------------------------------------------------------
// RMSNorm: out = x * rsqrt(mean(x^2)+eps) * w     (single block, 256 threads)
// ---------------------------------------------------------------------------
__global__ void rmsnorm_kernel(const float* __restrict__ x,
                               const float* __restrict__ w,
                               float* __restrict__ out, int n) {
  __shared__ float red[256];
  __shared__ float scale;
  float s = 0.f;
  for (int i = threadIdx.x; i < n; i += 256) { float v = x[i]; s += v * v; }
  red[threadIdx.x] = s;
  __syncthreads();
  for (int off = 128; off; off >>= 1) {
    if (threadIdx.x < off) red[threadIdx.x] += red[threadIdx.x + off];
    __syncthreads();
  }
  if (threadIdx.x == 0) scale = rsqrtf(red[0] / (float)n + 1e-5f);
  __syncthreads();
  for (int i = threadIdx.x; i < n; i += 256) out[i] = x[i] * scale * w[i];
}

// ---------------------------------------------------------------------------
// GEMV: y[row] = (res ? res[row] : 0) + dot(W[row,:], x)
// One wave per row, float4 streaming loads (bandwidth bound path).
// Block = 256 threads = 8 waves = 8 rows.
// ---------------------------------------------------------------------------
template <int COLS>
__global__ void gemv_kernel(const float* __restrict__ W,
                            const float* __restrict__ x,
                            const float* __restrict__ res,
                            float* __restrict__ y, int rows) {
  __shared__ float xs[COLS];
  const int tid = threadIdx.x;
  stage_x(x, xs, COLS / 4, tid);
  const int wave = tid >> 5, lane = tid & 31;
  const int row = blockIdx.x * 8 + wave;
  if (row >= rows) return;
  const float4* Wr = (const float4*)(W + (size_t)row * COLS);
  float acc = 0.f;
  #pragma unroll 4
  for (int i = lane; i < COLS / 4; i += 32) {
    __builtin_prefetch(Wr + i + 64, 0, 0);  // stream-ahead (speculative)
    float4 wv = Wr[i];
    float4 xv = ((float4*)xs)[i];
    acc += wv.x * xv.x + wv.y * xv.y + wv.z * xv.z + wv.w * xv.w;
  }
  acc = waveReduceSum(acc);
  if (lane == 0) y[row] = (res ? res[row] : 0.f) + acc;
}

// ---------------------------------------------------------------------------
// Fused SwiGLU gate: g[row] = silu(dot(W1[row],x)) * dot(W3[row],x)
// rows = 4096, cols = 1024
// ---------------------------------------------------------------------------
__global__ void ffn_gate_kernel(const float* __restrict__ W1,
                                const float* __restrict__ W3,
                                const float* __restrict__ x,
                                float* __restrict__ g) {
  __shared__ float xs[H];
  const int tid = threadIdx.x;
  stage_x(x, xs, H / 4, tid);
  const int wave = tid >> 5, lane = tid & 31;
  const int row = blockIdx.x * 8 + wave;
  const float4* r1 = (const float4*)(W1 + (size_t)row * H);
  const float4* r3 = (const float4*)(W3 + (size_t)row * H);
  float a1 = 0.f, a3 = 0.f;
  #pragma unroll 4
  for (int i = lane; i < H / 4; i += 32) {
    float4 xv = ((float4*)xs)[i];
    float4 w1v = r1[i], w3v = r3[i];
    a1 += w1v.x * xv.x + w1v.y * xv.y + w1v.z * xv.z + w1v.w * xv.w;
    a3 += w3v.x * xv.x + w3v.y * xv.y + w3v.z * xv.z + w3v.w * xv.w;
  }
  a1 = waveReduceSum(a1);
  a3 = waveReduceSum(a3);
  if (lane == 0) g[row] = (a1 / (1.f + __expf(-a1))) * a3;
}

// ---------------------------------------------------------------------------
// Per-head RMSNorm + RoPE for q/k; write k,v into KV cache at POS.
// 32 blocks of 64 threads: blocks 0..15 = q heads, 16..23 = k heads,
// 24..31 = v head copy.
// ---------------------------------------------------------------------------
__global__ void qk_prep_kernel(const float* __restrict__ qkv,
                               const float* __restrict__ q_ln,
                               const float* __restrict__ k_ln,
                               const int* __restrict__ posp,
                               float* __restrict__ qrot,
                               float* __restrict__ kv_out, int s) {
  const int b = blockIdx.x, d = threadIdx.x;
  const int pos = posp[0];
  if (b < 24) {
    __shared__ float xsh[HD];
    __shared__ float red[HD];
    __shared__ float scale;
    const int isQ = (b < 16);
    const int hh = isQ ? b : (b - 16);
    const float* src = isQ ? (qkv + hh * HD) : (qkv + H + hh * HD);
    const float* ln  = isQ ? q_ln : k_ln;
    const float v = src[d];
    red[d] = v * v;
    __syncthreads();
    for (int off = 32; off; off >>= 1) {
      if (d < off) red[d] += red[d + off];
      __syncthreads();
    }
    if (d == 0) scale = rsqrtf(red[0] / (float)HD + 1e-5f);
    __syncthreads();
    const float xn = v * scale * ln[d];
    xsh[d] = xn;
    __syncthreads();
    const float inv = powf(1000000.0f, -(float)(d & 31) / 32.0f);
    const float ang = (float)pos * inv;
    const float rot = (d < 32) ? -xsh[d + 32] : xsh[d - 32];
    const float o = xn * cosf(ang) + rot * sinf(ang);
    if (isQ) qrot[hh * HD + d] = o;
    else kv_out[(((size_t)s * NKV + hh) * CTX + pos) * HD + d] = o;
  } else {
    const int g = b - 24;
    kv_out[(((size_t)(NA + s) * NKV + g) * CTX + pos) * HD + d] =
        qkv[H + NKV * HD + g * HD + d];
  }
}

// ---------------------------------------------------------------------------
// Depthwise short-conv: shift register update + conv + gating.
// Writes the full padded new_state slot (output) each call.
// ---------------------------------------------------------------------------
__global__ void conv_kernel(const float* __restrict__ bcx,
                            const float* __restrict__ state_in,
                            const float* __restrict__ cw,
                            const float* __restrict__ cb,
                            float* __restrict__ slots_out,
                            float* __restrict__ y, int s) {
  const int c = blockIdx.x * blockDim.x + threadIdx.x;
  if (c >= H) return;
  const float B_ = bcx[c], C_ = bcx[H + c], xb = bcx[2 * H + c];
  const float Bx = B_ * xb;
  const float* sl = state_in + ((size_t)s * H + c) * 16;
  const float l0 = sl[1], l1 = sl[2], l2 = Bx;
  float* so = slots_out + ((size_t)s * H + c) * 16;
  so[0] = l0; so[1] = l1; so[2] = l2;
  #pragma unroll
  for (int j = 3; j < 16; j++) so[j] = 0.f;
  const float* w = cw + ((size_t)s * H + c) * 3;
  const float cout = l0 * w[0] + l1 * w[1] + l2 * w[2] + cb[s * H + c];
  y[c] = C_ * cout;
}

// ---------------------------------------------------------------------------
// WMMA attention: one block (4 waves) per KV head.
// Phase 1: each wave computes scores for a 512-key range with
//          v_wmma_f32_16x16x32_f16 (2 q-rows padded to 16).
// Phase 2: block-wide softmax over 2048 keys (LDS reductions).
// Phase 3: each wave owns one 16-column group of Out = P @ V (WMMA).
// A/B/C lane layouts per CDNA5 ISA §7.12.2; each WMMA runs on a
// full wave (EXEC all-ones).
// ---------------------------------------------------------------------------
__global__ void __launch_bounds__(128)
attn_kernel(const float* __restrict__ qrot,
            const float* __restrict__ kv,
            const int* __restrict__ posp,
            float* __restrict__ attout, int s) {
  const int g = blockIdx.x;
  const int tid = threadIdx.x;
  const int lane = tid & 31;
  const int w = tid >> 5;          // wave id 0..3
  const int pos = posp[0];
  __shared__ float sc[2][CTX];
  __shared__ float red0[128], red1[128];
  __shared__ float m0s, m1s, s0s, s1s;

  const float* K = kv + ((size_t)(s * NKV + g)) * CTX * HD;
  const float* V = kv + ((size_t)((NA + s) * NKV + g)) * CTX * HD;

  const int M  = lane & 15;        // A row / B,C column index for this lane
  const int hi = lane >> 4;        // lane half: selects K-pair group
  const int kbase = hi * 16;       // unused; B lane half holds k 8*hi..8*hi+7 and 16+8*hi..23+8*hi

  // A = Q tile (16x64 split into two 16x32), rows 0..1 = q heads 2g,2g+1
  v16h aLo, aHi;
  #pragma unroll
  for (int j = 0; j < 8; j++) {
    const int pb = (j < 4) ? (hi ? 8 + 2 * j : 2 * j)
                           : (hi ? 24 + 2 * (j - 4) : 16 + 2 * (j - 4));
    float q0 = 0.f, q1 = 0.f, q2 = 0.f, q3 = 0.f;
    if (M < 2) {
      const float* q = qrot + (2 * g + M) * HD;
      q0 = q[pb]; q1 = q[pb + 1]; q2 = q[32 + pb]; q3 = q[32 + pb + 1];
    }
    aLo[2 * j] = (_Float16)q0; aLo[2 * j + 1] = (_Float16)q1;
    aHi[2 * j] = (_Float16)q2; aHi[2 * j + 1] = (_Float16)q3;
  }

  // ---- Phase 1: scores; wave w handles keys [512*w, 512*w+512) ----
  const float scale = 0.125f;  // 1/sqrt(64)
  for (int t0 = 512 * w; t0 < 512 * w + 512; t0 += 16) {
    const int key = t0 + M;                    // B column -> key index
    const float* Krow = K + (size_t)key * HD;
    v16h bLo, bHi;
    #pragma unroll
    for (int e = 0; e < 16; e++) {
      bLo[e] = (_Float16)Krow[hi * 8 + e + (e & 8)];
      bHi[e] = (_Float16)Krow[32 + hi * 8 + e + (e & 8)];
    }
    v8f c = {};
    c = __builtin_amdgcn_wmma_f32_16x16x32_f16(false, aLo, false, bLo,
                                               (short)0, c, false, false);
    c = __builtin_amdgcn_wmma_f32_16x16x32_f16(false, aHi, false, bHi,
                                               (short)0, c, false, false);
    if (hi == 0) {  // rows 0,1 live in VGPR0/1 on lanes 0..15
      const float m = (key <= pos) ? 0.f : -1e9f;
      sc[0][key] = c[0] * scale + m;
      sc[1][key] = c[1] * scale + m;
    }
  }
  __syncthreads();

  // ---- Phase 2: block-wide softmax over 2048 keys for both rows ----
  float mx0 = -1e30f, mx1 = -1e30f;
  for (int t = tid; t < CTX; t += 128) {
    mx0 = fmaxf(mx0, sc[0][t]);
    mx1 = fmaxf(mx1, sc[1][t]);
  }
  red0[tid] = mx0; red1[tid] = mx1;
  __syncthreads();
  for (int off = 64; off; off >>= 1) {
    if (tid < off) {
      red0[tid] = fmaxf(red0[tid], red0[tid + off]);
      red1[tid] = fmaxf(red1[tid], red1[tid + off]);
    }
    __syncthreads();
  }
  if (tid == 0) { m0s = red0[0]; m1s = red1[0]; }
  __syncthreads();
  mx0 = m0s; mx1 = m1s;

  float sum0 = 0.f, sum1 = 0.f;
  for (int t = tid; t < CTX; t += 128) {
    const float p0 = __expf(sc[0][t] - mx0);
    const float p1 = __expf(sc[1][t] - mx1);
    sc[0][t] = p0; sc[1][t] = p1;
    sum0 += p0; sum1 += p1;
  }
  red0[tid] = sum0; red1[tid] = sum1;
  __syncthreads();
  for (int off = 64; off; off >>= 1) {
    if (tid < off) {
      red0[tid] += red0[tid + off];
      red1[tid] += red1[tid + off];
    }
    __syncthreads();
  }
  if (tid == 0) { s0s = red0[0]; s1s = red1[0]; }
  __syncthreads();

  // ---- Phase 3: Out = P @ V ; wave w owns columns [16*w, 16*w+16) ----
  const int n0 = 16 * w;
  v8f c = {};
  for (int t0 = 0; t0 < CTX; t0 += 32) {
    v16h a;
    #pragma unroll
    for (int j = 0; j < 8; j++) {
      const int pb = (j < 4) ? (hi ? 8 + 2 * j : 2 * j)
                             : (hi ? 24 + 2 * (j - 4) : 16 + 2 * (j - 4));
      float p0 = 0.f, p1 = 0.f;
      if (M < 2) { p0 = sc[M][t0 + pb]; p1 = sc[M][t0 + pb + 1]; }
      a[2 * j] = (_Float16)p0; a[2 * j + 1] = (_Float16)p1;
    }
    v16h b;
    #pragma unroll
    for (int e = 0; e < 16; e++) {
      b[e] = (_Float16)V[(size_t)(t0 + hi * 8 + e + (e & 8)) * HD + n0 + M];
    }
    c = __builtin_amdgcn_wmma_f32_16x16x32_f16(false, a, false, b,
                                               (short)0, c, false, false);
  }
  if (hi == 0) {
    attout[(2 * g + 0) * HD + n0 + M] = c[0] / s0s;
    attout[(2 * g + 1) * HD + n0 + M] = c[1] / s1s;
  }
}

// ---------------------------------------------------------------------------
// Host orchestration
// ---------------------------------------------------------------------------
extern "C" void kernel_launch(void* const* d_in, const int* in_sizes, int n_in,
                              void* d_out, int out_size, void* d_ws, size_t ws_size,
                              hipStream_t stream) {
  const int*   ids        = (const int*)  d_in[0];
  const int*   posp       = (const int*)  d_in[1];
  const float* conv_state = (const float*)d_in[4];
  const float* kv_in      = (const float*)d_in[5];
  const float* embed_w    = (const float*)d_in[6];
  const float* lm_head_w  = (const float*)d_in[7];
  const float* emb_norm   = (const float*)d_in[8];
  const float* op_norm    = (const float*)d_in[9];
  const float* ffn_norm   = (const float*)d_in[10];
  const float* q_w        = (const float*)d_in[11];
  const float* k_w        = (const float*)d_in[12];
  const float* v_w        = (const float*)d_in[13];
  const float* o_w        = (const float*)d_in[14];
  const float* q_ln       = (const float*)d_in[15];
  const float* k_ln       = (const float*)d_in[16];
  const float* in_proj    = (const float*)d_in[17];
  const float* conv_w     = (const float*)d_in[18];
  const float* conv_b     = (const float*)d_in[19];
  const float* out_proj   = (const float*)d_in[20];
  const float* w1         = (const float*)d_in[21];
  const float* w3         = (const float*)d_in[22];
  const float* w2         = (const float*)d_in[23];

  float* logits    = (float*)d_out;
  float* kv_out    = logits + VOCAB;
  float* slots_out = kv_out + (size_t)2 * NA * NKV * CTX * HD;

  float* ws     = (float*)d_ws;
  float* hidden = ws;            // 1024
  float* xn     = ws + 1024;     // 1024
  float* qkv    = ws + 2048;     // 2048
  float* qrot   = ws + 4096;     // 1024
  float* attout = ws + 5120;     // 1024
  float* bcx    = ws + 6144;     // 3072
  float* convy  = ws + 9216;     // 1024
  float* gate   = ws + 10240;    // 4096

  // kv_cache output = input cache with row POS overwritten (never mutate input)
  (void)hipMemcpyAsync(kv_out, kv_in,
                       (size_t)2 * NA * NKV * CTX * HD * sizeof(float),
                       hipMemcpyDeviceToDevice, stream);

  embed_kernel<<<4, 256, 0, stream>>>(ids, embed_w, hidden);

  static const int is_attn[NL] = {0,0,1,0,0,1,0,0,1,0,0,1};
  int aslot = 0, cslot = 0;
  for (int li = 0; li < NL; li++) {
    rmsnorm_kernel<<<1, 256, 0, stream>>>(hidden, op_norm + li * H, xn, H);
    if (is_attn[li]) {
      const int s = aslot++;
      gemv_kernel<H><<<128, 256, 0, stream>>>(q_w + (size_t)s * NH * HD * H, xn,
                                              nullptr, qkv, NH * HD);
      gemv_kernel<H><<<64, 256, 0, stream>>>(k_w + (size_t)s * NKV * HD * H, xn,
                                             nullptr, qkv + H, NKV * HD);
      gemv_kernel<H><<<64, 256, 0, stream>>>(v_w + (size_t)s * NKV * HD * H, xn,
                                             nullptr, qkv + H + NKV * HD, NKV * HD);
      qk_prep_kernel<<<32, 64, 0, stream>>>(qkv, q_ln + s * HD, k_ln + s * HD,
                                            posp, qrot, kv_out, s);
      attn_kernel<<<NKV, 128, 0, stream>>>(qrot, kv_out, posp, attout, s);
      gemv_kernel<H><<<128, 256, 0, stream>>>(o_w + (size_t)s * H * NH * HD,
                                              attout, hidden, hidden, H);
    } else {
      const int s = cslot++;
      gemv_kernel<H><<<384, 256, 0, stream>>>(in_proj + (size_t)s * 3 * H * H, xn,
                                              nullptr, bcx, 3 * H);
      conv_kernel<<<4, 256, 0, stream>>>(bcx, conv_state, conv_w, conv_b,
                                         slots_out, convy, s);
      gemv_kernel<H><<<128, 256, 0, stream>>>(out_proj + (size_t)s * H * H,
                                              convy, hidden, hidden, H);
    }
    rmsnorm_kernel<<<1, 256, 0, stream>>>(hidden, ffn_norm + li * H, xn, H);
    ffn_gate_kernel<<<512, 256, 0, stream>>>(w1 + (size_t)li * DFF * H,
                                             w3 + (size_t)li * DFF * H, xn, gate);
    gemv_kernel<DFF><<<128, 256, 0, stream>>>(w2 + (size_t)li * H * DFF, gate,
                                              hidden, hidden, H);
  }

  rmsnorm_kernel<<<1, 256, 0, stream>>>(hidden, emb_norm, xn, H);
  gemv_kernel<H><<<4096, 256, 0, stream>>>(lm_head_w, xn, nullptr, logits, VOCAB);
}
